// TorchSPN_89051851915508
// MI455X (gfx1250) — matrix operand order repaired and checked by reading, not compiled, not run on hardware
//
#include <hip/hip_runtime.h>

typedef __attribute__((ext_vector_type(16))) _Float16 v16h;
typedef __attribute__((ext_vector_type(8)))  float    v8f;

#define HALF_LOG_2PI 0.9189385332046727f
#define LOG2E        1.4426950408889634f
#define LN2          0.6931471805599453f

__device__ __forceinline__ float4 ld4(const float* p) { return *(const float4*)p; }

// ---------------------------------------------------------------------------
// Prep 1: leaf parameters.  h(x) = a*(x-mu)^2 + off
//   a = -0.5*exp(-2*log_sigma), off = -log_sigma - 0.5*log(2*pi)
// ---------------------------------------------------------------------------
extern "C" __global__ void spn_prep_leaf(const float* __restrict__ mu,
                                         const float* __restrict__ ls,
                                         float* __restrict__ leafA,
                                         float* __restrict__ leafMu,
                                         float* __restrict__ leafOff) {
  int tid = blockIdx.x * blockDim.x + threadIdx.x;
  if (tid >= 2048) return;
  float l = ls[tid];
  leafA[tid]   = -0.5f * __builtin_amdgcn_exp2f(-2.0f * l * LOG2E);
  leafMu[tid]  = mu[tid];
  leafOff[tid] = -l - HALF_LOG_2PI;
}

// ---------------------------------------------------------------------------
// Prep 2: softmax weights over c (axis=1 of [R,64,8]) and pre-swizzle into the
// WMMA f16 B operand (32x16, 8 VGPRs) in this code's contiguous k order, same as A:
//   lane = N + 16*(K>=16), element t = K&15 within each K-half block (hb).
// 1024 halves per global region: [hb(2)][lane(32)][t(16)].  Columns N=8..15 = 0.
// ---------------------------------------------------------------------------
extern "C" __global__ void spn_prep_w(const float* __restrict__ w,
                                      _Float16* __restrict__ wsB,
                                      int R, int gbase) {
  int tid = blockIdx.x * blockDim.x + threadIdx.x;
  if (tid >= R * 8) return;
  int r = tid >> 3, k = tid & 7;
  const float* wr = w + (size_t)r * 512;           // [64][8]
  float mx = wr[k];
  for (int c = 1; c < 64; ++c) mx = fmaxf(mx, wr[c * 8 + k]);
  float e[64], s = 0.f;
  for (int c = 0; c < 64; ++c) {
    e[c] = __builtin_amdgcn_exp2f((wr[c * 8 + k] - mx) * LOG2E);
    s += e[c];
  }
  float inv = 1.0f / s;
  _Float16* dst = wsB + (size_t)(gbase + r) * 1024;
  for (int c = 0; c < 64; ++c) {
    int hb = c >> 5, kc = c & 31;
    int lsub = (kc >> 4) & 1, t = kc & 15;
    int lane = k + 16 * lsub;
    dst[(hb * 32 + lane) * 16 + t]     = (_Float16)(e[c] * inv);  // column N=k
    dst[(hb * 32 + lane + 8) * 16 + t] = (_Float16)0.0f;          // column N=k+8 pad
  }
}

// ---------------------------------------------------------------------------
// One product+sum layer step for a 16-row batch tile owned by one wave.
// src[m*sm + slot*8 + k] (f32 log-probs, slots pair 2r,2r+1) ->
// dst[m*dm + r*8 + k].
// mm is 32 floats: parity-double-buffered per region to decouple the DS chain.
// ---------------------------------------------------------------------------
__device__ __forceinline__ void layer_step(const float* src, int sm,
                                           float* dst, int dm, int nreg,
                                           const _Float16* __restrict__ wsB,
                                           int gbase, float* mm, int lane) {
  const int M  = lane & 15;
  const int hi = lane >> 4;
#pragma unroll 2
  for (int r = 0; r < nreg; ++r) {
    float* mms = mm + 16 * (r & 1);
    const float* c0 = src + M * sm + (2 * r) * 8;
    float4 q0 = ld4(c0), q1 = ld4(c0 + 4), q2 = ld4(c0 + 8), q3 = ld4(c0 + 12);
    float h0[8] = {q0.x, q0.y, q0.z, q0.w, q1.x, q1.y, q1.z, q1.w};
    float h1[8] = {q2.x, q2.y, q2.z, q2.w, q3.x, q3.y, q3.z, q3.w};
    float mx0 = h0[0], mx1 = h1[0];
#pragma unroll
    for (int j = 1; j < 8; ++j) { mx0 = fmaxf(mx0, h0[j]); mx1 = fmaxf(mx1, h1[j]); }
    float mmax = mx0 + mx1;                 // exact max over c of h0_i + h1_j
    if (hi == 0) mms[M] = mmax;             // stash for D-layout phase

    // pre-scaled, max-shifted children: one v_add + one v_exp per A element
    float a0[8], a1[8];
#pragma unroll
    for (int j = 0; j < 8; ++j) {
      a0[j] = (h0[j] - mx0) * LOG2E;
      a1[j] = (h1[j] - mx1) * LOG2E;
    }

    // A operand, this code's contiguous k order (matches B): element t in lane L holds
    // K = t + 16*(L>=16), M = L&15.
    v16h A1, A2;
#pragma unroll
    for (int t = 0; t < 16; ++t) {
      int c  = t + (hi ? 16 : 0);
      int c2 = c + 32;
      A1[t] = (_Float16)__builtin_amdgcn_exp2f(a0[c  >> 3] + a1[c  & 7]);
      A2[t] = (_Float16)__builtin_amdgcn_exp2f(a0[c2 >> 3] + a1[c2 & 7]);
    }
    const _Float16* wp = wsB + (size_t)(gbase + r) * 1024;
    v16h B1 = *(const v16h*)(wp + lane * 16);        // K = 0..31 block
    v16h B2 = *(const v16h*)(wp + 512 + lane * 16);  // K = 32..63 block
    v8f C = {};
    C = __builtin_amdgcn_wmma_f32_16x16x32_f16(false, A1, false, B1,
                                               (short)0, C, false, false);
    C = __builtin_amdgcn_wmma_f32_16x16x32_f16(false, A2, false, B2,
                                               (short)0, C, false, false);
    // D layout: element i holds M = i + 8*(lane>=16), N = lane&15 (valid N<8)
    // s never denormal (max-shifted e contains 1.0): raw v_log_f32 is safe.
    if (M < 8) {
      float4 mA = ld4(mms + 8 * hi);        // two ds_load_b128, batched
      float4 mB = ld4(mms + 8 * hi + 4);
      float mrow[8] = {mA.x, mA.y, mA.z, mA.w, mB.x, mB.y, mB.z, mB.w};
#pragma unroll
      for (int i = 0; i < 8; ++i) {
        int m = i + 8 * hi;
        dst[m * dm + r * 8 + M] = __builtin_amdgcn_logf(C[i]) * LN2 + mrow[i];
      }
    }
  }
}

// ---------------------------------------------------------------------------
// Main kernel: 4 waves/block, 16 batch rows per wave, depth-first 16-feature
// chunks.  All intermediates wave-private in LDS (5152 floats/wave).
// ---------------------------------------------------------------------------
extern "C" __global__ __launch_bounds__(128)
void spn_main(const float* __restrict__ x, const _Float16* __restrict__ wsB,
              const float* __restrict__ leafA, const float* __restrict__ leafMu,
              const float* __restrict__ leafOff, const float* __restrict__ rootw,
              float* __restrict__ out) {
  extern __shared__ float smem[];
  const int lane = threadIdx.x & 31;
  const int wv   = threadIdx.x >> 5;
  const int M    = lane & 15;
  const int hi   = lane >> 4;
  float* base = smem + wv * 5152;
  float* bufL = base;          // [16 m][16 slot][8 k]
  float* buf0 = base + 2048;   // [16 m][ 8 slot][8 k]
  float* h3   = base + 3072;   // [16 m][16 slot][8 k]
  float* mm   = base + 5120;   // [2][16] parity buffers
  const int b0 = blockIdx.x * 64 + wv * 16;

  for (int chunk = 0; chunk < 16; ++chunk) {
    // leaf log-probs for 16 features of this chunk (lane halves split features)
    const float* xrow = x + (size_t)(b0 + M) * 256 + chunk * 16 + hi * 8;
    float4 xa = ld4(xrow), xb = ld4(xrow + 4);
    float xv[8] = {xa.x, xa.y, xa.z, xa.w, xb.x, xb.y, xb.z, xb.w};
#pragma unroll
    for (int ff = 0; ff < 8; ++ff) {
      int fl = hi * 8 + ff;
      int f  = chunk * 16 + fl;
      float4 pA0 = ld4(leafA  + f * 8), pA1 = ld4(leafA  + f * 8 + 4);
      float4 pM0 = ld4(leafMu + f * 8), pM1 = ld4(leafMu + f * 8 + 4);
      float4 pO0 = ld4(leafOff + f * 8), pO1 = ld4(leafOff + f * 8 + 4);
      float av[8] = {pA0.x, pA0.y, pA0.z, pA0.w, pA1.x, pA1.y, pA1.z, pA1.w};
      float mv[8] = {pM0.x, pM0.y, pM0.z, pM0.w, pM1.x, pM1.y, pM1.z, pM1.w};
      float ov[8] = {pO0.x, pO0.y, pO0.z, pO0.w, pO1.x, pO1.y, pO1.z, pO1.w};
      float hv[8];
#pragma unroll
      for (int k = 0; k < 8; ++k) {
        float d = xv[ff] - mv[k];
        hv[k] = fmaf(av[k] * d, d, ov[k]);
      }
      float* dp = bufL + M * 128 + fl * 8;
      *(float4*)dp       = make_float4(hv[0], hv[1], hv[2], hv[3]);
      *(float4*)(dp + 4) = make_float4(hv[4], hv[5], hv[6], hv[7]);
    }
    layer_step(bufL, 128, buf0, 64, 8, wsB,        chunk * 8, mm, lane);  // L0
    layer_step(buf0,  64, bufL, 128, 4, wsB, 128 + chunk * 4, mm, lane);  // L1
    layer_step(bufL, 128, buf0, 64, 2, wsB, 192 + chunk * 2, mm, lane);   // L2
    layer_step(buf0,  64, h3 + chunk * 8, 128, 1, wsB, 224 + chunk, mm, lane); // L3
  }
  layer_step(h3,   128, bufL, 128, 8, wsB, 240, mm, lane);  // L4
  layer_step(bufL, 128, buf0,  64, 4, wsB, 248, mm, lane);  // L5
  layer_step(buf0,  64, bufL, 128, 2, wsB, 252, mm, lane);  // L6
  layer_step(bufL, 128, buf0,  64, 1, wsB, 254, mm, lane);  // L7

  // root: lp = h7[m,0,:] + log_softmax(root_w); out = logsumexp(lp)
  if (hi == 0) {
    float mx = rootw[0];
#pragma unroll
    for (int k = 1; k < 8; ++k) mx = fmaxf(mx, rootw[k]);
    float s = 0.f;
#pragma unroll
    for (int k = 0; k < 8; ++k) s += __builtin_amdgcn_exp2f((rootw[k] - mx) * LOG2E);
    float lz = __builtin_amdgcn_logf(s) * LN2 + mx;
    float4 hA = ld4(buf0 + M * 64), hB = ld4(buf0 + M * 64 + 4);
    float hv[8] = {hA.x, hA.y, hA.z, hA.w, hB.x, hB.y, hB.z, hB.w};
    float lp[8], best = -3.4e38f;
#pragma unroll
    for (int k = 0; k < 8; ++k) {
      lp[k] = hv[k] + (rootw[k] - lz);
      best = fmaxf(best, lp[k]);
    }
    float ss = 0.f;
#pragma unroll
    for (int k = 0; k < 8; ++k) ss += __builtin_amdgcn_exp2f((lp[k] - best) * LOG2E);
    out[b0 + M] = __builtin_amdgcn_logf(ss) * LN2 + best;
  }
}

// ---------------------------------------------------------------------------
extern "C" void kernel_launch(void* const* d_in, const int* in_sizes, int n_in,
                              void* d_out, int out_size, void* d_ws, size_t ws_size,
                              hipStream_t stream) {
  (void)in_sizes; (void)n_in; (void)out_size; (void)ws_size;
  const float* x      = (const float*)d_in[0];
  const float* mu     = (const float*)d_in[1];
  const float* ls     = (const float*)d_in[2];
  const float* root_w = (const float*)d_in[11];
  float* out = (float*)d_out;

  // workspace: 255 regions * 1024 halves f16 = 522240 B, then 3*2048 f32 leaf params
  _Float16* wsB   = (_Float16*)d_ws;
  float* leafA    = (float*)((char*)d_ws + 255 * 2048);
  float* leafMu   = leafA + 2048;
  float* leafOff  = leafMu + 2048;

  spn_prep_leaf<<<8, 256, 0, stream>>>(mu, ls, leafA, leafMu, leafOff);

  int R = 128, gbase = 0;
  for (int l = 0; l < 8; ++l) {
    const float* w = (const float*)d_in[3 + l];
    int n = R * 8;
    spn_prep_w<<<(n + 127) / 128, 128, 0, stream>>>(w, wsB, R, gbase);
    gbase += R;
    R >>= 1;
  }

  // 128 blocks * 4 waves * 16 rows = 8192 batch rows; 82432 B dynamic LDS
  spn_main<<<128, 128, 4 * 5152 * 4, stream>>>(x, wsB, leafA, leafMu, leafOff,
                                               root_w, out);
}
